// ODERNNCell_80041010528506
// MI455X (gfx1250) — compile-verified, matrix operand order checked by reading, not run on hardware
//
#include <hip/hip_runtime.h>
#include <math.h>

typedef __attribute__((ext_vector_type(16))) __bf16 v16bf;
typedef __attribute__((ext_vector_type(8)))  float  v8f;

#define B_TOTAL 131072
#define E_DIM   128
#define H_DIM   128
#define D_DIM   256
#define LN_EPS  1e-5f

#define NW   2        // waves per block (wave32)
#define STR  260      // LDS stage row stride in floats (260 % 64 == 4 -> conflict-free column reads)

// ---- weight fragment buffer layout (bf16, in d_ws) ----
// region elems = NT * KC * 32 lanes * 16 bf16
#define W1_OFF   0            // NT=16, KC=4  -> 32768
#define W2_OFF   32768        // NT=8,  KC=8  -> 32768
#define WIH_OFF  65536        // NT=24, KC=4  -> 49152
#define WHH_OFF  114688       // NT=24, KC=4  -> 49152
#define WTOTAL   163840

// A/B fragment element (j, lane) -> (k within 32-chunk, n within 16-tile)
__device__ __forceinline__ void frag_kn(int lane, int j, int& krel, int& nrel) {
    int v = j >> 1;
    krel = ((v < 4) ? (2 * v) : (16 + 2 * (v - 4))) + ((lane >= 16) ? 8 : 0) + (j & 1);
    nrel = lane & 15;
}

// ------------------------- weight prep: f32 -> bf16 pre-swizzled B fragments -------------------------
__global__ void prep_weights(const float* __restrict__ w1, const float* __restrict__ w2,
                             const float* __restrict__ wih, const float* __restrict__ whh,
                             __bf16* __restrict__ out) {
    int i = blockIdx.x * blockDim.x + threadIdx.x;
    if (i >= WTOTAL) return;

    int rem, KC; const float* src; int mode; // mode 0: B[k][n]=src[k*N+n]; 1: B[k][n]=src[n*K+k]
    int Ncols;
    if (i < W2_OFF)       { rem = i - W1_OFF;  KC = 4; src = w1;  mode = 0; Ncols = D_DIM; }
    else if (i < WIH_OFF) { rem = i - W2_OFF;  KC = 8; src = w2;  mode = 0; Ncols = H_DIM; }
    else if (i < WHH_OFF) { rem = i - WIH_OFF; KC = 4; src = wih; mode = 1; Ncols = E_DIM; }
    else                  { rem = i - WHH_OFF; KC = 4; src = whh; mode = 1; Ncols = H_DIM; }

    int j    = rem & 15;
    int lane = (rem >> 4) & 31;
    int tc   = rem >> 9;
    int c    = tc % KC;
    int t    = tc / KC;

    int krel, nrel;
    frag_kn(lane, j, krel, nrel);
    int k = 32 * c + krel;
    int n = 16 * t + nrel;

    float v = (mode == 0) ? src[k * Ncols + n]   // w1:(H,D) / w2:(D,H) used as K x N
                          : src[n * Ncols + k];  // gru weights (3H,E)/(3H,H) used transposed
    out[i] = (__bf16)v;
}

// ------------------------- device helpers -------------------------
__device__ __forceinline__ v16bf load_bfrag(const __bf16* base, int t, int c, int KC, int lane) {
    return *(const v16bf*)(base + ((((t * KC + c) * 32 + lane) << 4)));
}

template <int KC>
__device__ __forceinline__ void build_afrags(const float* sh, int lane, v16bf* out) {
    int M = lane & 15;
    int hio = (lane >= 16) ? 8 : 0;
#pragma unroll
    for (int c = 0; c < KC; ++c) {
        v16bf f;
#pragma unroll
        for (int j = 0; j < 16; ++j) {
            int v = j >> 1;
            int krel = ((v < 4) ? (2 * v) : (16 + 2 * (v - 4))) + hio + (j & 1);
            f[j] = (__bf16)sh[M * STR + 32 * c + krel];
        }
        out[c] = f;
    }
}

// coalesced global(16x128 tile) <-> LDS stage
__device__ __forceinline__ void stage_from_global128(const float* g, float* sh, int lane) {
    for (int idx = lane; idx < 16 * 128; idx += 32) {
        int r = idx >> 7, cc = idx & 127;
        sh[r * STR + cc] = g[idx];
    }
}
__device__ __forceinline__ void stage_to_global128(float* g, const float* sh, int lane) {
    for (int idx = lane; idx < 16 * 128; idx += 32) {
        int r = idx >> 7, cc = idx & 127;
        g[idx] = sh[r * STR + cc];
    }
}

__device__ __forceinline__ void cfrags_to_stage(const v8f* fr, int nfr, float* sh, int lane) {
    int col0 = lane & 15;
    int hi = lane >> 4;
    for (int t = 0; t < nfr; ++t)
#pragma unroll
        for (int r = 0; r < 8; ++r)
            sh[(r + 8 * hi) * STR + 16 * t + col0] = fr[t][r];
}

// dynamics f: arg (16xH f32) must already be staged in sh[0:16][0:128].
// Produces kout[8] C-fragments = Linear -> LayerNorm -> GELU(exact) -> Linear.
__device__ void dynamics_f(float* sh, float* red2, int lane,
                           const __bf16* w1f, const __bf16* w2f,
                           const float* __restrict__ b1, const float* __restrict__ lnw,
                           const float* __restrict__ lnb, const float* __restrict__ b2,
                           v8f* kout) {
    int col0 = lane & 15;
    int hi = lane >> 4;
    int M = col0;

    __syncthreads();                       // staged arg visible
    v16bf A1[4];
    build_afrags<4>(sh, lane, A1);
    __syncthreads();                       // all lanes done reading arg

    // GEMM1: u = arg @ w1 + b1, write u tiles (16x256 f32) into stage
    for (int t = 0; t < 16; ++t) {
        v8f acc = {};
#pragma unroll
        for (int c = 0; c < 4; ++c) {
            v16bf bf = load_bfrag(w1f, t, c, 4, lane);
            acc = __builtin_amdgcn_wmma_f32_16x16x32_bf16(false, A1[c], false, bf,
                                                          (short)0, acc, false, false);
        }
        float bias = b1[16 * t + col0];
#pragma unroll
        for (int r = 0; r < 8; ++r)
            sh[(r + 8 * hi) * STR + 16 * t + col0] = acc[r] + bias;
    }
    __syncthreads();

    // LayerNorm over D=256 per row, then exact GELU, in place (f32)
    {
        float s = 0.f, sq = 0.f;
        int cbase = 128 * hi;
        for (int cc = 0; cc < 128; ++cc) {
            float v = sh[M * STR + cbase + cc];
            s += v; sq += v * v;
        }
        red2[(hi * 16 + M) * 2 + 0] = s;
        red2[(hi * 16 + M) * 2 + 1] = sq;
        __syncthreads();
        float st  = red2[M * 2 + 0] + red2[(16 + M) * 2 + 0];
        float sqt = red2[M * 2 + 1] + red2[(16 + M) * 2 + 1];
        float mu  = st * (1.f / 256.f);
        float var = sqt * (1.f / 256.f) - mu * mu;
        float rstd = rsqrtf(var + LN_EPS);
        for (int cc = 0; cc < 128; ++cc) {
            int col = cbase + cc;
            float v = (sh[M * STR + col] - mu) * rstd * lnw[col] + lnb[col];
            v = 0.5f * v * (1.f + erff(v * 0.70710678118654752f));
            sh[M * STR + col] = v;
        }
    }
    __syncthreads();

    // GEMM2: k = a @ w2 + b2  (K = 256)
    v16bf A2[8];
    build_afrags<8>(sh, lane, A2);
    for (int t = 0; t < 8; ++t) {
        v8f acc = {};
#pragma unroll
        for (int c = 0; c < 8; ++c) {
            v16bf bf = load_bfrag(w2f, t, c, 8, lane);
            acc = __builtin_amdgcn_wmma_f32_16x16x32_bf16(false, A2[c], false, bf,
                                                          (short)0, acc, false, false);
        }
        float bias = b2[16 * t + col0];
#pragma unroll
        for (int r = 0; r < 8; ++r) acc[r] += bias;
        kout[t] = acc;
    }
}

// ------------------------- main fused ODE-RNN kernel -------------------------
__global__ __launch_bounds__(NW * 32)
void odernn_kernel(const float* __restrict__ h_in, const float* __restrict__ dts,
                   const float* __restrict__ obs, const int* __restrict__ mask,
                   const float* __restrict__ b1, const float* __restrict__ lnw,
                   const float* __restrict__ lnb, const float* __restrict__ b2,
                   const float* __restrict__ bih, const float* __restrict__ bhh,
                   const __bf16* __restrict__ wfrag,
                   float* __restrict__ out_evolved, float* __restrict__ out_updated) {
    __shared__ float stage_s[NW][16 * STR];
    __shared__ float red_s[NW][64];

    int wave = threadIdx.x >> 5;
    int lane = threadIdx.x & 31;
    float* sh   = stage_s[wave];
    float* red2 = red_s[wave];
    int row0 = (blockIdx.x * NW + wave) * 16;
    int col0 = lane & 15;
    int hi = lane >> 4;

    const __bf16* w1f  = wfrag + W1_OFF;
    const __bf16* w2f  = wfrag + W2_OFF;
    const __bf16* wihf = wfrag + WIH_OFF;
    const __bf16* whhf = wfrag + WHH_OFF;

    // warm GRU weights toward L2 while RK4 runs (global_prefetch_b8)
    for (int p = lane; p < 384; p += 32)
        __builtin_prefetch((const void*)(wihf + (size_t)p * 256), 0, 1);

    // per-lane row scalars (row = row0 + r + 8*hi for fragment register r)
    float dtv[8];
    int   mrow[8];
#pragma unroll
    for (int r = 0; r < 8; ++r) {
        int row = row0 + r + 8 * hi;
        float d = dts[row];
        dtv[r] = d > 0.f ? d : 0.f;
        mrow[r] = mask[row];
    }

    // load h tile (coalesced) and pick up C-fragments
    stage_from_global128(h_in + (size_t)row0 * H_DIM, sh, lane);
    __syncthreads();
    v8f hfr[8];
#pragma unroll
    for (int t = 0; t < 8; ++t) {
        v8f f;
#pragma unroll
        for (int r = 0; r < 8; ++r) f[r] = sh[(r + 8 * hi) * STR + 16 * t + col0];
        hfr[t] = f;
    }

    // ---- RK4 (3/8 rule): stage already holds h for k1 ----
    v8f k1[8], k2[8], cur[8], acc[8];
    dynamics_f(sh, red2, lane, w1f, w2f, b1, lnw, lnb, b2, k1);
#pragma unroll
    for (int t = 0; t < 8; ++t) {
        acc[t] = k1[t];
#pragma unroll
        for (int r = 0; r < 8; ++r)
            cur[t][r] = hfr[t][r] + dtv[r] * (k1[t][r] * (1.f / 3.f));
    }
    __syncthreads();
    cfrags_to_stage(cur, 8, sh, lane);
    dynamics_f(sh, red2, lane, w1f, w2f, b1, lnw, lnb, b2, k2);
#pragma unroll
    for (int t = 0; t < 8; ++t)
#pragma unroll
        for (int r = 0; r < 8; ++r) {
            acc[t][r] += 3.f * k2[t][r];
            cur[t][r] = hfr[t][r] + dtv[r] * (k2[t][r] - k1[t][r] * (1.f / 3.f));
        }
    __syncthreads();
    cfrags_to_stage(cur, 8, sh, lane);
    dynamics_f(sh, red2, lane, w1f, w2f, b1, lnw, lnb, b2, cur);  // cur = k3
#pragma unroll
    for (int t = 0; t < 8; ++t)
#pragma unroll
        for (int r = 0; r < 8; ++r) {
            acc[t][r] += 3.f * cur[t][r];
            cur[t][r] = hfr[t][r] + dtv[r] * (k1[t][r] - k2[t][r] + cur[t][r]);
        }
    __syncthreads();
    cfrags_to_stage(cur, 8, sh, lane);
    dynamics_f(sh, red2, lane, w1f, w2f, b1, lnw, lnb, b2, cur);  // cur = k4

    v8f ev[8];
#pragma unroll
    for (int t = 0; t < 8; ++t)
#pragma unroll
        for (int r = 0; r < 8; ++r)
            ev[t][r] = hfr[t][r] + dtv[r] * ((acc[t][r] + cur[t][r]) * 0.125f);
    // dt<=0 samples: dtv==0 -> ev == h exactly (matches where(dt>0, stepped, h))

    // store evolved + build its A-fragments from the same staged copy
    __syncthreads();
    cfrags_to_stage(ev, 8, sh, lane);
    __syncthreads();
    stage_to_global128(out_evolved + (size_t)row0 * H_DIM, sh, lane);
    v16bf Ah[4];
    build_afrags<4>(sh, lane, Ah);

    // observation embedding A-fragments
    __syncthreads();
    stage_from_global128(obs + (size_t)row0 * E_DIM, sh, lane);
    __syncthreads();
    v16bf Ax[4];
    build_afrags<4>(sh, lane, Ax);
    __syncthreads();

    // ---- GRU update, per H-column tile j ----
    v8f upd[8];
#pragma unroll
    for (int j = 0; j < 8; ++j) {
        v8f ir = {}, iz = {}, in_ = {}, hr = {}, hz = {}, hn = {};
#pragma unroll
        for (int c = 0; c < 4; ++c) {
            ir  = __builtin_amdgcn_wmma_f32_16x16x32_bf16(false, Ax[c], false, load_bfrag(wihf, j,      c, 4, lane), (short)0, ir,  false, false);
            iz  = __builtin_amdgcn_wmma_f32_16x16x32_bf16(false, Ax[c], false, load_bfrag(wihf, j + 8,  c, 4, lane), (short)0, iz,  false, false);
            in_ = __builtin_amdgcn_wmma_f32_16x16x32_bf16(false, Ax[c], false, load_bfrag(wihf, j + 16, c, 4, lane), (short)0, in_, false, false);
            hr  = __builtin_amdgcn_wmma_f32_16x16x32_bf16(false, Ah[c], false, load_bfrag(whhf, j,      c, 4, lane), (short)0, hr,  false, false);
            hz  = __builtin_amdgcn_wmma_f32_16x16x32_bf16(false, Ah[c], false, load_bfrag(whhf, j + 8,  c, 4, lane), (short)0, hz,  false, false);
            hn  = __builtin_amdgcn_wmma_f32_16x16x32_bf16(false, Ah[c], false, load_bfrag(whhf, j + 16, c, 4, lane), (short)0, hn,  false, false);
        }
        float bir = bih[      16 * j + col0], biz = bih[128 + 16 * j + col0], bin = bih[256 + 16 * j + col0];
        float bhr = bhh[      16 * j + col0], bhz = bhh[128 + 16 * j + col0], bhn = bhh[256 + 16 * j + col0];
#pragma unroll
        for (int r = 0; r < 8; ++r) {
            float rr = 1.f / (1.f + expf(-(ir[r] + bir + hr[r] + bhr)));
            float zz = 1.f / (1.f + expf(-(iz[r] + biz + hz[r] + bhz)));
            float nn = tanhf(in_[r] + bin + rr * (hn[r] + bhn));
            float go = (1.f - zz) * nn + zz * ev[j][r];
            upd[j][r] = (mrow[r] > 0) ? go : ev[j][r];
        }
    }

    __syncthreads();
    cfrags_to_stage(upd, 8, sh, lane);
    __syncthreads();
    stage_to_global128(out_updated + (size_t)row0 * H_DIM, sh, lane);
}

// ------------------------- launch -------------------------
extern "C" void kernel_launch(void* const* d_in, const int* in_sizes, int n_in,
                              void* d_out, int out_size, void* d_ws, size_t ws_size,
                              hipStream_t stream) {
    const float* h   = (const float*)d_in[0];
    const float* dt  = (const float*)d_in[1];
    const float* obs = (const float*)d_in[2];
    const int*   mk  = (const int*)d_in[3];
    const float* w1  = (const float*)d_in[4];
    const float* b1  = (const float*)d_in[5];
    const float* lnw = (const float*)d_in[6];
    const float* lnb = (const float*)d_in[7];
    const float* w2  = (const float*)d_in[8];
    const float* b2  = (const float*)d_in[9];
    const float* wih = (const float*)d_in[10];
    const float* whh = (const float*)d_in[11];
    const float* bih = (const float*)d_in[12];
    const float* bhh = (const float*)d_in[13];

    __bf16* wf = (__bf16*)d_ws;
    float* oev = (float*)d_out;
    float* oup = oev + (size_t)B_TOTAL * H_DIM;

    prep_weights<<<(WTOTAL + 255) / 256, 256, 0, stream>>>(w1, w2, wih, whh, wf);
    odernn_kernel<<<B_TOTAL / (16 * NW), NW * 32, 0, stream>>>(
        h, dt, obs, mk, b1, lnw, lnb, b2, bih, bhh, wf, oev, oup);
}
